// Code2VecEncoder_11115375362278
// MI455X (gfx1250) — matrix operand order repaired and checked by reading, compiled, not run on hardware
//
#include <hip/hip_runtime.h>
#include <stdint.h>

// Problem constants (from reference)
#define B_   512
#define L_   200
#define LP   208      // L padded to 13*16
#define E_   128
#define K3E  384      // 3*E, GEMM K
#define D_   384      // GEMM N
#define LT   13       // L tiles of 16
#define DT   24       // D tiles of 16
#define KT   12       // K tiles of 32
#define RS   392      // padded ctx row stride in ushorts (784B = 196 dwords == 4 mod 64 banks)
#define NTHREADS 256
#define NWAVES   8

typedef __attribute__((ext_vector_type(16))) __bf16 v16bf;
typedef __attribute__((ext_vector_type(8)))  float  v8f;

union FragAB {
    v16bf bf;
    uint4 q[2];
};

__device__ __forceinline__ unsigned short f2bf(float f) {
    unsigned int u = __float_as_uint(f);
    u += 0x7FFFu + ((u >> 16) & 1u);      // round-to-nearest-even f32 -> bf16
    return (unsigned short)(u >> 16);
}

__device__ __forceinline__ float fast_tanh(float x) {
#if __has_builtin(__builtin_amdgcn_tanhf)
    return __builtin_amdgcn_tanhf(x);     // v_tanh_f32 (CDNA5 TRANS op)
#elif __has_builtin(__builtin_amdgcn_tanh_f32)
    return __builtin_amdgcn_tanh_f32(x);
#else
    float t = __expf(-2.0f * __builtin_fabsf(x));
    float r = (1.0f - t) / (1.0f + t);
    return __builtin_copysignf(r, x);
#endif
}

// decode 8 bf16 packed in a uint4 to f32
__device__ __forceinline__ void bfdec8(uint4 q, float f[8]) {
    f[0] = __uint_as_float(q.x << 16); f[1] = __uint_as_float(q.x & 0xffff0000u);
    f[2] = __uint_as_float(q.y << 16); f[3] = __uint_as_float(q.y & 0xffff0000u);
    f[4] = __uint_as_float(q.z << 16); f[5] = __uint_as_float(q.z & 0xffff0000u);
    f[6] = __uint_as_float(q.w << 16); f[7] = __uint_as_float(q.w & 0xffff0000u);
}

// Pre-convert W_fc [D, 3E] f32 -> bf16 into workspace (288 KB, stays in L2/L0)
__global__ void w_to_bf16_kernel(const float* __restrict__ w,
                                 unsigned short* __restrict__ o, int n) {
    int i = blockIdx.x * blockDim.x + threadIdx.x;
    if (i < n) o[i] = f2bf(w[i]);
}

__global__ __launch_bounds__(NTHREADS)
void code2vec_kernel(const int* __restrict__ xs1, const int* __restrict__ p1,
                     const int* __restrict__ xt1,
                     const int* __restrict__ xs2, const int* __restrict__ p2,
                     const int* __restrict__ xt2,
                     const float* __restrict__ word_emb,
                     const float* __restrict__ path_emb,
                     const unsigned short* __restrict__ wbf,
                     const float* __restrict__ attn,
                     float* __restrict__ out) {
    // ctx rows: 384 data ushorts + 8 hole ushorts (16B). Holes host sc[] (rows 0-51)
    // and cv[] (rows 52-147). GEMM/h traffic only touches cols < 384.
    __shared__ __align__(16) unsigned short ctx[LP * RS];  // 163,072 B

#define SCP(l) ((float*)(ctx + ((l) >> 2) * RS + 384) + ((l) & 3))
#define CVP(n) ((float*)(ctx + (52 + ((n) >> 2)) * RS + 384) + ((n) & 3))

    const int b    = blockIdx.x;
    const int e    = blockIdx.y;
    const int tid  = threadIdx.x;
    const int lane = tid & 31;
    const int wave = tid >> 5;
    const int half = lane >> 4;   // 0: lanes 0-15, 1: lanes 16-31
    const int nIdx = lane & 15;

    const int* xs = e ? xs2 : xs1;
    const int* pp = e ? p2  : p1;
    const int* xt = e ? xt2 : xt1;

    // zero cv accumulator (lives in row holes; untouched by col<384 traffic)
    for (int i = tid; i < D_; i += NTHREADS) *CVP(i) = 0.f;

    // ---- gather + f32->bf16 into LDS (float4-vectorized; segments = 32 quads) ----
    for (int q = tid; q < L_ * (K3E / 4); q += NTHREADS) {
        const int l  = q / (K3E / 4);
        const int cq = q - l * (K3E / 4);
        const float4* src;
        if (cq < 32)      src = (const float4*)&word_emb[(size_t)xs[b * L_ + l] * E_] + cq;
        else if (cq < 64) src = (const float4*)&path_emb[(size_t)pp[b * L_ + l] * E_] + (cq - 32);
        else              src = (const float4*)&word_emb[(size_t)xt[b * L_ + l] * E_] + (cq - 64);
        const float4 v = *src;
        ushort4 o;
        o.x = f2bf(v.x); o.y = f2bf(v.y); o.z = f2bf(v.z); o.w = f2bf(v.w);
        *(ushort4*)&ctx[l * RS + cq * 4] = o;
    }
    // zero data cols of pad rows 200..207 (holes there unused)
    for (int q = tid; q < 8 * (K3E / 4); q += NTHREADS) {
        const int l  = L_ + q / (K3E / 4);
        const int cq = q % (K3E / 4);
        *(ushort4*)&ctx[l * RS + cq * 4] = ushort4{0, 0, 0, 0};
    }
    __syncthreads();

    // ==== single GEMM pass: h = tanh(ctx @ W^T); h overwrites ctx row-block lt ====
    // wave owns dt in {wave, wave+8, wave+16}. Per lt: all waves read rows lt,
    // barrier, then write tanh'd bf16 h back into the same rows.
    for (int lt = 0; lt < LT; ++lt) {
        unsigned short tp[3][8];
#pragma unroll
        for (int j = 0; j < 3; ++j) {
            const int dt = wave + j * 8;
            v8f c = {};
#pragma unroll 4
            for (int kt = 0; kt < KT; ++kt) {
                FragAB a, bm;
                // A 16x32 bf16: half 0 holds K 0-7 & 16-23, half 1 holds 8-15 & 24-31
                const int m  = lt * 16 + nIdx;
                const int k0 = kt * 32 + half * 8;
                a.q[0] = *(const uint4*)&ctx[m * RS + k0];
                a.q[1] = *(const uint4*)&ctx[m * RS + k0 + 16];
                // B 32x16 bf16: W^T[k][n] = W[n][k]; half 0 holds K 0-7 & 16-23, half 1 holds 8-15 & 24-31 of row n
                const int n  = dt * 16 + nIdx;
                const int kb = kt * 32 + half * 8;
                const uint4* wq = (const uint4*)&wbf[(size_t)n * K3E + kb];
                bm.q[0] = wq[0];
                bm.q[1] = wq[2];
                c = __builtin_amdgcn_wmma_f32_16x16x32_bf16(
                        false, a.bf, false, bm.bf, (short)0, c, false, false);
            }
#pragma unroll
            for (int r = 0; r < 8; ++r)
                tp[j][r] = f2bf(fast_tanh(c[r]));
        }
        __syncthreads();   // all waves done READING rows lt
#pragma unroll
        for (int j = 0; j < 3; ++j) {
            const int dt = wave + j * 8;
#pragma unroll
            for (int r = 0; r < 8; ++r)   // C/D layout: M = r + 8*half, N = nIdx
                ctx[(lt * 16 + r + half * 8) * RS + dt * 16 + nIdx] = tp[j][r];
        }
    }
    __syncthreads();       // h fully resident in ctx

    // ---- scores[l] = sum_n h[l][n] * attn[n]  (row per thread; attn is uniform->SMEM)
    if (tid < LP) {
        float acc = 0.f;
        for (int o = 0; o < 48; ++o) {
            const uint4 q = *(const uint4*)&ctx[tid * RS + o * 8];
            float f[8];
            bfdec8(q, f);
            const float4 a0 = *(const float4*)&attn[o * 8];
            const float4 a1 = *(const float4*)&attn[o * 8 + 4];
            acc += f[0] * a0.x + f[1] * a0.y + f[2] * a0.z + f[3] * a0.w
                 + f[4] * a1.x + f[5] * a1.y + f[6] * a1.z + f[7] * a1.w;
        }
        *SCP(tid) = acc;   // pad rows: h==0 -> score 0 (masked in softmax anyway)
    }
    __syncthreads();

    // ---- softmax over L (single wave)
    if (wave == 0) {
        float mx = -3.402823466e38f;
        for (int l = lane; l < LP; l += 32)
            mx = fmaxf(mx, (l < L_) ? *SCP(l) : -3.402823466e38f);
        for (int off = 16; off; off >>= 1) mx = fmaxf(mx, __shfl_xor(mx, off));
        float sum = 0.f;
        for (int l = lane; l < LP; l += 32) {
            float ev = (l < L_) ? __expf(*SCP(l) - mx) : 0.f;
            *SCP(l) = ev;
            sum += ev;
        }
        for (int off = 16; off; off >>= 1) sum += __shfl_xor(sum, off);
        const float inv = 1.f / sum;
        for (int l = lane; l < LP; l += 32) *SCP(l) *= inv;
    }
    __syncthreads();

    // ---- code_vec[n] = sum_l a[l] * h[l][n] : 48 col-octs x 4 row-chunks of 52
    if (tid < 192) {
        const int oct   = tid % 48;
        const int chunk = tid / 48;
        float acc[8];
#pragma unroll
        for (int j = 0; j < 8; ++j) acc[j] = 0.f;
        for (int r = 0; r < 52; ++r) {
            const int row = chunk * 52 + r;
            const float a = *SCP(row);
            const uint4 q = *(const uint4*)&ctx[row * RS + oct * 8];
            float f[8];
            bfdec8(q, f);
#pragma unroll
            for (int j = 0; j < 8; ++j) acc[j] += a * f[j];
        }
#pragma unroll
        for (int j = 0; j < 8; ++j)
            atomicAdd(CVP(oct * 8 + j), acc[j]);   // ds_add_f32
    }
    __syncthreads();

    // ---- write code_vec: d_out = [cv1 (512x384) | cv2 (512x384)]
    float* ob = out + ((size_t)e * B_ + b) * D_;
    for (int d = tid; d < D_; d += NTHREADS) ob[d] = *CVP(d);

#undef SCP
#undef CVP
}

extern "C" void kernel_launch(void* const* d_in, const int* in_sizes, int n_in,
                              void* d_out, int out_size, void* d_ws, size_t ws_size,
                              hipStream_t stream) {
    const int*   xs1      = (const int*)d_in[0];
    const int*   p1       = (const int*)d_in[1];
    const int*   xt1      = (const int*)d_in[2];
    const int*   xs2      = (const int*)d_in[3];
    const int*   p2       = (const int*)d_in[4];
    const int*   xt2      = (const int*)d_in[5];
    const float* word_emb = (const float*)d_in[6];
    const float* path_emb = (const float*)d_in[7];
    const float* w_fc     = (const float*)d_in[8];
    const float* attn     = (const float*)d_in[9];

    unsigned short* wbf = (unsigned short*)d_ws;   // needs D_*K3E*2 = 294,912 B

    const int nW = D_ * K3E;
    hipLaunchKernelGGL(w_to_bf16_kernel, dim3((nW + 255) / 256), dim3(256), 0, stream,
                       w_fc, wbf, nW);
    hipLaunchKernelGGL(code2vec_kernel, dim3(B_, 2), dim3(NTHREADS), 0, stream,
                       xs1, p1, xt1, xs2, p2, xt2,
                       word_emb, path_emb, wbf, attn, (float*)d_out);
}
